// GCN_23476291240112
// MI455X (gfx1250) — matrix operand order repaired and checked by reading, not compiled, not run on hardware
//
#include <hip/hip_runtime.h>
#include <hip/hip_bf16.h>

#define NROW 8192
#define INC  128
#define HIDC 256
#define OUTC 128

typedef __bf16 bf16;
typedef __attribute__((ext_vector_type(8)))  __bf16 v8bf;
typedef __attribute__((ext_vector_type(16))) __bf16 v16bf;
typedef __attribute__((ext_vector_type(8)))  float  v8f;

__device__ __forceinline__ float sigmoidf_fast(float v) {
    return 1.0f / (1.0f + __expf(-v));
}

// ---------------------------------------------------------------------------
// K0: zero the degree accumulator.
// ---------------------------------------------------------------------------
__global__ void k_zero(float* __restrict__ p, int n) {
    int i = blockIdx.x * 256 + threadIdx.x;
    if (i < n) p[i] = 0.0f;
}

// ---------------------------------------------------------------------------
// K1: symmetric tile pass over the UPPER triangle of the 8192x8192 graph.
// A = sigmoid(I + 0.5*(P+P^T)) is symmetric, so each 64x64 tile (bi<=bj) is
// computed once, written to both mirror positions of unnormalized
// A_hat = A + I (bf16), and contributes degree partial-sums to both the
// row band (bi) and column band (bj) via global f32 atomics.
// P is read exactly once in total.
// ---------------------------------------------------------------------------
__global__ void k_sym_tiles(const float* __restrict__ P, float* __restrict__ deg,
                            bf16* __restrict__ Ahat) {
    const int bi = blockIdx.y, bj = blockIdx.x;
    if (bj < bi) return;                 // upper triangle only (uniform exit)
    __shared__ float T2[64][65];         // P[bj-band, bi-band] for transposed read
    __shared__ float S[64][65];          // sigmoid tile
    const int t  = threadIdx.x;
    const int i0 = bi * 64, j0 = bj * 64;

    #pragma unroll
    for (int s = 0; s < 16; ++s) {
        int idx = t + 256 * s;
        int row = idx >> 6, col = idx & 63;
        T2[row][col] = P[(size_t)(j0 + row) * NROW + (i0 + col)];
    }
    __syncthreads();

    #pragma unroll
    for (int s = 0; s < 16; ++s) {
        int idx = t + 256 * s;
        int ii = idx >> 6, jj = idx & 63;
        int i = i0 + ii, j = j0 + jj;
        float pij = P[(size_t)i * NROW + j];
        float v = 0.5f * (pij + T2[jj][ii]);
        if (i == j) v += 1.0f;           // +I inside sigmoid argument (sym)
        S[ii][jj] = sigmoidf_fast(v);
    }
    __syncthreads();

    // write A_hat[i][j] = S + I  (coalesced along jj)
    #pragma unroll
    for (int s = 0; s < 16; ++s) {
        int idx = t + 256 * s;
        int ii = idx >> 6, jj = idx & 63;
        int i = i0 + ii, j = j0 + jj;
        float a = S[ii][jj] + ((i == j) ? 1.0f : 0.0f);
        Ahat[(size_t)i * NROW + j] = (bf16)a;
    }
    // mirror write A_hat[j][i] (off-diagonal blocks; coalesced along ii)
    if (bj != bi) {
        #pragma unroll
        for (int s = 0; s < 16; ++s) {
            int idx = t + 256 * s;
            int jj = idx >> 6, ii = idx & 63;
            Ahat[(size_t)(j0 + jj) * NROW + (i0 + ii)] = (bf16)S[ii][jj];
        }
    }

    // degree contributions: rows of S -> band bi, cols of S -> band bj
    if (t < 64) {
        float rs = 0.0f;
        #pragma unroll
        for (int jj = 0; jj < 64; ++jj) rs += S[t][jj];
        atomicAdd(deg + i0 + t, rs);
    } else if (t < 128 && bj != bi) {
        int jj = t - 64;
        float cs = 0.0f;
        #pragma unroll
        for (int ii = 0; ii < 64; ++ii) cs += S[ii][jj];
        atomicAdd(deg + j0 + jj, cs);
    }
}

// ---------------------------------------------------------------------------
// K2: dinv = rsqrt(deg + 1)   (+1 = self loop of A_hat)
// ---------------------------------------------------------------------------
__global__ void k_dinv(const float* __restrict__ deg, float* __restrict__ dinv) {
    int i = blockIdx.x * 256 + threadIdx.x;
    dinv[i] = rsqrtf(deg[i] + 1.0f);
}

// ---------------------------------------------------------------------------
// K3: in-place symmetric normalization of bf16 A_hat: A[i][j] *= dinv_i*dinv_j
// 8 contiguous elements per thread (b128 load/store).
// ---------------------------------------------------------------------------
__global__ void k_scale_A(bf16* __restrict__ Abf, const float* __restrict__ dinv) {
    size_t base = ((size_t)blockIdx.x * 256 + threadIdx.x) * 8;
    int i = (int)(base >> 13);           // / 8192
    int j = (int)(base & 8191);
    float di = dinv[i];
    v8bf v = *(v8bf*)(Abf + base);
    #pragma unroll
    for (int k = 0; k < 8; ++k)
        v[k] = (bf16)((float)v[k] * di * dinv[j + k]);
    *(v8bf*)(Abf + base) = v;
}

// ---------------------------------------------------------------------------
// K4: XW1T[h][n] = (x @ W1)^T in bf16 (transposed so GEMM B-fragments are
// contiguous per-lane 32B loads).
// ---------------------------------------------------------------------------
__global__ void k_xw1t(const float* __restrict__ x, const float* __restrict__ W1,
                       bf16* __restrict__ XW1T) {
    const int n = blockIdx.x * 256 + threadIdx.x;
    const int h = blockIdx.y;
    float acc = 0.0f;
    #pragma unroll 4
    for (int k = 0; k < INC; ++k)
        acc += x[(size_t)n * INC + k] * W1[(size_t)k * HIDC + h];
    XW1T[(size_t)h * NROW + n] = (bf16)acc;
}

// ---------------------------------------------------------------------------
// K6: HW2T[o][n] = (h @ W2)^T in bf16.
// ---------------------------------------------------------------------------
__global__ void k_hw2t(const bf16* __restrict__ hmat, const float* __restrict__ W2,
                       bf16* __restrict__ HW2T) {
    const int n = blockIdx.x * 256 + threadIdx.x;
    const int o = blockIdx.y;
    float acc = 0.0f;
    #pragma unroll 4
    for (int k = 0; k < HIDC; ++k)
        acc += (float)hmat[(size_t)n * HIDC + k] * W2[(size_t)k * OUTC + o];
    HW2T[(size_t)o * NROW + n] = (bf16)acc;
}

// ---------------------------------------------------------------------------
// WMMA GEMM: out[M x Nn] = A[M x 8192] (bf16 row-major) * BT[Nn x 8192]^T
//            + bias, optional ReLU. Wave computes a 16 x (NTILES*16) strip.
// A-fragment (16-bit A 16x32): lane<16 holds row, K0..15;
//   lane>=16 K16..31 -> two contiguous 16B loads per lane.
// B-fragment: lane<16 = column n=lane K0..15; lane>=16 K16..31
//   -> one contiguous 32B load per lane from the transposed B.
// Branch-free inner loop so the compiler clause-schedules b128 loads against
// in-flight WMMAs (s_wait_loadcnt 0x1 pipelining).
// ---------------------------------------------------------------------------
template<int NTILES, bool RELU, bool OUT_BF16>
__global__ void k_gemm_wmma(const bf16* __restrict__ A, const bf16* __restrict__ BT,
                            const float* __restrict__ bias, void* __restrict__ out,
                            int Nn) {
    const int lane  = threadIdx.x & 31;
    const int wave  = threadIdx.x >> 5;
    const int mBase = (blockIdx.x * 8 + wave) * 16;
    const int nBase = blockIdx.y * (NTILES * 16);
    const int l15   = lane & 15;
    const int hi    = (lane >= 16) ? 1 : 0;
    const int aKoff = hi * 16;
    const int bKoff = hi * 16;

    v8f acc[NTILES] = {};
    const bf16* aPtr = A + (size_t)(mBase + l15) * NROW;

    for (int kk = 0; kk < NROW; kk += 32) {
        v8bf alo = *(const v8bf*)(aPtr + kk + aKoff);
        v8bf ahi = *(const v8bf*)(aPtr + kk + aKoff + 8);
        v16bf afrag = __builtin_shufflevector(alo, ahi,
            0, 1, 2, 3, 4, 5, 6, 7, 8, 9, 10, 11, 12, 13, 14, 15);
        #pragma unroll
        for (int nt = 0; nt < NTILES; ++nt) {
            int col = nBase + nt * 16 + l15;
            v16bf bfrag = *(const v16bf*)(BT + (size_t)col * NROW + kk + bKoff);
            acc[nt] = __builtin_amdgcn_wmma_f32_16x16x32_bf16(
                false, afrag, false, bfrag, (short)0, acc[nt], false, false);
        }
    }

    // C/D layout: VGPR v -> M = mBase + v + 8*hi, N = nBase + nt*16 + l15
    #pragma unroll
    for (int nt = 0; nt < NTILES; ++nt) {
        int col = nBase + nt * 16 + l15;
        float b = bias[col];
        #pragma unroll
        for (int v = 0; v < 8; ++v) {
            int row = mBase + v + 8 * hi;
            float val = acc[nt][v] + b;
            if (RELU) val = fmaxf(val, 0.0f);
            if (OUT_BF16)
                ((bf16*)out)[(size_t)row * Nn + col] = (bf16)val;
            else
                ((float*)out)[(size_t)row * Nn + col] = val;
        }
    }
}

// ---------------------------------------------------------------------------
extern "C" void kernel_launch(void* const* d_in, const int* in_sizes, int n_in,
                              void* d_out, int out_size, void* d_ws, size_t ws_size,
                              hipStream_t stream) {
    const float* x  = (const float*)d_in[0];
    const float* P  = (const float*)d_in[1];
    const float* W1 = (const float*)d_in[2];
    const float* b1 = (const float*)d_in[3];
    const float* W2 = (const float*)d_in[4];
    const float* b2 = (const float*)d_in[5];

    char* ws = (char*)d_ws;
    size_t off = 0;
    bf16* Abf   = (bf16*)(ws + off); off += (size_t)NROW * NROW * sizeof(bf16); // 128 MB
    bf16* XW1T  = (bf16*)(ws + off); off += (size_t)HIDC * NROW * sizeof(bf16); //   4 MB
    bf16* HW2T  = (bf16*)(ws + off); off += (size_t)OUTC * NROW * sizeof(bf16); //   2 MB
    bf16* Hbuf  = (bf16*)(ws + off); off += (size_t)NROW * HIDC * sizeof(bf16); //   4 MB
    float* dinv = (float*)(ws + off); off += (size_t)NROW * sizeof(float);      //  32 KB
    float* deg  = (float*)(ws + off);                                           //  32 KB

    // adjacency pipeline (P read exactly once)
    k_zero<<<NROW / 256, 256, 0, stream>>>(deg, NROW);
    k_sym_tiles<<<dim3(NROW / 64, NROW / 64), 256, 0, stream>>>(P, deg, Abf);
    k_dinv<<<NROW / 256, 256, 0, stream>>>(deg, dinv);
    k_scale_A<<<(NROW / 256) * (NROW / 8), 256, 0, stream>>>(Abf, dinv);
    // (x @ W1)^T  (independent of adjacency)
    k_xw1t<<<dim3(NROW / 256, HIDC), 256, 0, stream>>>(x, W1, XW1T);
    // h = relu(A @ xW1 + b1)   [WMMA, bf16 out]
    k_gemm_wmma<4, true, true><<<dim3(NROW / 128, HIDC / 64), 256, 0, stream>>>(
        Abf, XW1T, b1, (void*)Hbuf, HIDC);
    // (h @ W2)^T
    k_hw2t<<<dim3(NROW / 256, OUTC), 256, 0, stream>>>(Hbuf, W2, HW2T);
    // out = A @ hW2 + b2       [WMMA, f32 out]
    k_gemm_wmma<4, false, false><<<dim3(NROW / 128, OUTC / 64), 256, 0, stream>>>(
        Abf, HW2T, b2, d_out, OUTC);
}
